// DetoxXlnetClassifier_44555990728843
// MI455X (gfx1250) — matrix operand order repaired and checked by reading, not compiled, not run on hardware
//
#include <hip/hip_runtime.h>
#include <math.h>
#include <stdint.h>

// ---------------- model dims ----------------
#define LNUM 12
#define D    768
#define H    12
#define DH   64
#define FF   3072
#define BB   8
#define Q    512
#define NT   (BB*Q)          // 4096 tokens
#define NLC  2
#define KRROWS 1056          // 1024 + zero padding (bd slab can touch row 1024)
#define SCALE_C 0.125f       // 1/sqrt(64)

typedef __attribute__((ext_vector_type(16))) __bf16 v16bf;
typedef __attribute__((ext_vector_type(8)))  __bf16 v8bf;
typedef __attribute__((ext_vector_type(8)))  float  v8f;
typedef __attribute__((ext_vector_type(4)))  unsigned int u32x4;
typedef __attribute__((ext_vector_type(8)))  int i32x8;
typedef __attribute__((ext_vector_type(4)))  int i32x4;

#define GF_F32  1
#define GF_BF16 2
#define GF_BIAS 4
#define GF_GELU 8

// GEMM tiling
#define TM 64
#define TN 128
#define TK 32
#define LDSPAD 40            // 32 elems + 8 pad (80B rows: conflict-free b128 reads)

// ---------------- WMMA fragment loaders ----------------
// A (16-bit, 16x32): lanes 0-15 = rows, K chunks [h*8, h*8+8) and [16+h*8, ...)
__device__ __forceinline__ v16bf load_fragA(const __bf16* base, long rowStride,
                                            int row0, int colBase, int lane) {
    int half = lane >> 4;
    int m    = lane & 15;
    const __bf16* p = base + (long)(row0 + m) * rowStride + colBase;
    union { v16bf v; v8bf h[2]; } f;
    f.h[0] = *(const v8bf*)(p + half * 8);
    f.h[1] = *(const v8bf*)(p + 16 + half * 8);
    return f.v;
}
// B (16-bit, 32x16) given as B^T rows: lane = column n, K chunks [h*8, h*8+8) and [16+h*8, ...)
__device__ __forceinline__ v16bf load_fragB(const __bf16* baseT, long rowStride,
                                            int col0, int kBase, int lane) {
    int half = lane >> 4;
    int nl   = lane & 15;
    const __bf16* p = baseT + (long)(col0 + nl) * rowStride + kBase;
    union { v16bf v; v8bf h[2]; } f;
    f.h[0] = *(const v8bf*)(p + half * 8);
    f.h[1] = *(const v8bf*)(p + 16 + half * 8);
    return f.v;
}

__device__ __forceinline__ float block_reduce(float v, float* red, int isMax) {
    int tid = threadIdx.x;
    red[tid] = v; __syncthreads();
    for (int s = 128; s > 0; s >>= 1) {
        if (tid < s) red[tid] = isMax ? fmaxf(red[tid], red[tid + s])
                                      : red[tid] + red[tid + s];
        __syncthreads();
    }
    float r = red[0]; __syncthreads();
    return r;
}

// ---------------- TDM: 2-D tile (rows x 32 bf16) global -> LDS ----------------
// D# group layouts per CDNA5 ISA ch.8. pad: 4 DWORDs after every 16 DWORDs
// -> LDS row stride 80B (LDSPAD elems). Issued by one wave; TENSORcnt tracked.
__device__ __forceinline__ void tdm_tile_load(uint32_t ldsOff, const __bf16* gsrc,
                                              uint32_t rows, uint32_t strideElems) {
    uint64_t ga = (uint64_t)(uintptr_t)gsrc;
    u32x4 g0;
    g0[0] = 1u;                                        // count=1, user descriptor
    g0[1] = ldsOff;                                    // lds_addr (bytes)
    g0[2] = (uint32_t)ga;                              // global_addr[31:0]
    g0[3] = (uint32_t)((ga >> 32) & 0x01FFFFFFu)       // global_addr[56:32]
            | (2u << 30);                              // type=2 ("image")
    i32x8 g1;
    g1[0] = (int)((1u << 16)                           // data_size = 2 bytes
            | (1u << 20)                               // pad_enable
            | (3u << 22)                               // pad_interval: 16 DWORDs
            | (3u << 25));                             // pad_amount: 4 DWORDs
    g1[1] = (int)((32u & 0xFFFFu) << 16);              // tensor_dim0 = 32 (lo16)
    g1[2] = (int)((rows & 0xFFFFu) << 16);             // dim0 hi=0 | tensor_dim1 lo16
    g1[3] = (int)(32u << 16);                          // dim1 hi=0 | tile_dim0 = 32
    g1[4] = (int)(rows & 0xFFFFu);                     // tile_dim1 = rows, tile_dim2=0
    g1[5] = (int)strideElems;                          // tensor_dim0_stride lo32
    g1[6] = 0;                                         // stride hi16 | dim1_stride lo16
    g1[7] = 0;
    i32x4 gz4 = {0, 0, 0, 0};
    i32x8 gz8 = {0, 0, 0, 0, 0, 0, 0, 0};
    // 6-arg form on this toolchain (clang-23 / therock-10.0 headers)
    __builtin_amdgcn_tensor_load_to_lds(g0, g1, gz4, gz4, gz8, 0);
}

// ---------------- conversion kernels ----------------
__global__ void k_conv(const float* __restrict__ src, __bf16* __restrict__ dst, long n) {
    long i = (long)blockIdx.x * blockDim.x + threadIdx.x;
    long stride = (long)gridDim.x * blockDim.x;
    for (; i < n; i += stride) dst[i] = (__bf16)src[i];
}

// src layer z: [R][C] f32  ->  dst layer z: [C][R] bf16
__global__ void k_convT(const float* __restrict__ src, __bf16* __restrict__ dst,
                        int R, int C, long srcStride, long dstStride) {
    __shared__ float tile[32][33];
    const float* s = src + (long)blockIdx.z * srcStride;
    __bf16*      d = dst + (long)blockIdx.z * dstStride;
    int c0 = blockIdx.x * 32, r0 = blockIdx.y * 32;
    for (int i = threadIdx.y; i < 32; i += 8) {
        int r = r0 + i, c = c0 + threadIdx.x;
        tile[i][threadIdx.x] = (r < R && c < C) ? s[(long)r * C + c] : 0.f;
    }
    __syncthreads();
    for (int i = threadIdx.y; i < 32; i += 8) {
        int c = c0 + i, r = r0 + threadIdx.x;
        if (c < C && r < R) d[(long)c * R + r] = (__bf16)tile[threadIdx.x][i];
    }
}

// ---------------- embedding + positional encoding ----------------
__global__ void k_embed(const int* __restrict__ tox, const float* __restrict__ emb,
                        float* __restrict__ h, __bf16* __restrict__ hbf) {
    int t = blockIdx.x;                 // t = i*BB + b
    int i = t / BB, b = t % BB;
    int id = tox[b * Q + i];
    const float* src = emb + (long)id * D;
    for (int c = threadIdx.x; c < D; c += blockDim.x) {
        float v = src[c];
        h[(long)t * D + c]   = v;
        hbf[(long)t * D + c] = (__bf16)v;
    }
}

__global__ void k_posenc(__bf16* __restrict__ rbf) {
    int j = blockIdx.x;                 // 0..1023, pos = 512 - j
    float pos = (float)(Q - j);
    const float lg = 9.2103403719761836f; // ln(10000)
    for (int c = threadIdx.x; c < D; c += blockDim.x) {
        int k = (c < D / 2) ? c : c - D / 2;
        float invf = expf(-(2.f * (float)k / (float)D) * lg);
        float s = pos * invf;
        float v = (c < D / 2) ? sinf(s) : cosf(s);
        rbf[(long)j * D + c] = (__bf16)v;
    }
}

// ---------------- TDM-staged bf16 WMMA GEMM ----------------
// C[M,N] = A[M,K] * B[K,N]; Bt = B^T rows. Block tile 64x128, 8 waves (2x4),
// each wave owns a 32x32 footprint (4 accumulators). Double-buffered LDS tiles
// filled by the Tensor Data Mover (wave 0 issues, s_wait_tensorcnt + barrier).
__global__ __launch_bounds__(256) void k_gemm(
    const __bf16* __restrict__ A, long lda,
    const __bf16* __restrict__ Bt, long ldb,
    float* __restrict__ C, __bf16* __restrict__ Cbf, long ldc,
    int M, int N, int K, const float* __restrict__ bias, int flags)
{
    __shared__ __align__(16) __bf16 Abuf[2][TM * LDSPAD];
    __shared__ __align__(16) __bf16 Bbuf[2][TN * LDSPAD];
    int lane = threadIdx.x & 31;
    int w    = threadIdx.x >> 5;
    int wi   = w >> 2;                 // 0..1
    int wj   = w & 3;                  // 0..3
    const __bf16* Ag = A  + (long)blockIdx.y * TM * lda;
    const __bf16* Bg = Bt + (long)blockIdx.x * TN * ldb;

    v8f acc[2][2];
#pragma unroll
    for (int ii = 0; ii < 2; ++ii)
#pragma unroll
        for (int jj = 0; jj < 2; ++jj)
            acc[ii][jj] = (v8f){0.f,0.f,0.f,0.f,0.f,0.f,0.f,0.f};

    int nchunk = K / TK;
    if (threadIdx.x < 32) {            // wave 0: prologue DMA for chunk 0
        tdm_tile_load((uint32_t)(uintptr_t)&Abuf[0][0], Ag, TM, (uint32_t)lda);
        tdm_tile_load((uint32_t)(uintptr_t)&Bbuf[0][0], Bg, TN, (uint32_t)ldb);
    }
    for (int c = 0; c < nchunk; ++c) {
        int buf = c & 1;
        if (threadIdx.x < 32) __builtin_amdgcn_s_wait_tensorcnt(0);
        __syncthreads();               // tile c visible to all waves
        if (threadIdx.x < 32 && (c + 1) < nchunk) {
            int nb = (c + 1) & 1;
            tdm_tile_load((uint32_t)(uintptr_t)&Abuf[nb][0], Ag + (c + 1) * TK, TM, (uint32_t)lda);
            tdm_tile_load((uint32_t)(uintptr_t)&Bbuf[nb][0], Bg + (c + 1) * TK, TN, (uint32_t)ldb);
        }
        const __bf16* Ab = &Abuf[buf][0];
        const __bf16* Bb = &Bbuf[buf][0];
        v16bf a0 = load_fragA(Ab, LDSPAD, wi * 32,      0, lane);
        v16bf a1 = load_fragA(Ab, LDSPAD, wi * 32 + 16, 0, lane);
        v16bf b0 = load_fragB(Bb, LDSPAD, wj * 32,      0, lane);
        v16bf b1 = load_fragB(Bb, LDSPAD, wj * 32 + 16, 0, lane);
        acc[0][0] = __builtin_amdgcn_wmma_f32_16x16x32_bf16(false, a0, false, b0,
                                                            (short)0, acc[0][0], false, false);
        acc[0][1] = __builtin_amdgcn_wmma_f32_16x16x32_bf16(false, a0, false, b1,
                                                            (short)0, acc[0][1], false, false);
        acc[1][0] = __builtin_amdgcn_wmma_f32_16x16x32_bf16(false, a1, false, b0,
                                                            (short)0, acc[1][0], false, false);
        acc[1][1] = __builtin_amdgcn_wmma_f32_16x16x32_bf16(false, a1, false, b1,
                                                            (short)0, acc[1][1], false, false);
        __syncthreads();               // all reads of tile c done before reuse
    }

    int lane15 = lane & 15, halfl = lane >> 4;
#pragma unroll
    for (int ii = 0; ii < 2; ++ii)
#pragma unroll
        for (int jj = 0; jj < 2; ++jj) {
            int col   = blockIdx.x * TN + wj * 32 + jj * 16 + lane15;
            int rbase = blockIdx.y * TM + wi * 32 + ii * 16 + halfl * 8;
            float bia = (flags & GF_BIAS) ? bias[col] : 0.f;
#pragma unroll
            for (int r = 0; r < 8; ++r) {
                float v = acc[ii][jj][r] + bia;
                if (flags & GF_GELU) v = 0.5f * v * (1.f + erff(v * 0.70710678118654752f));
                long o = (long)(rbase + r) * ldc + col;
                if (flags & GF_F32)  C[o]   = v;
                if (flags & GF_BF16) Cbf[o] = (__bf16)v;
            }
        }
}

// ---------------- q bias add (two biased bf16 copies) ----------------
__global__ void k_qbias(const float* __restrict__ qh,
                        const float* __restrict__ rwb, const float* __restrict__ rrb,
                        __bf16* __restrict__ qac, __bf16* __restrict__ qbd) {
    long i = (long)blockIdx.x * blockDim.x + threadIdx.x;
    if (i >= (long)NT * D) return;
    int c = (int)(i % D);
    float v = qh[i];
    qac[i] = (__bf16)(v + rwb[c]);
    qbd[i] = (__bf16)(v + rrb[c]);
}

// ---------------- attention scores: ac + rel-shifted bd ----------------
// score[b,n,i,j] = SCALE*((qac_i . kh_j) + (qbd_i . kr_{j-i+512})) - 1e30*mask
__global__ __launch_bounds__(256) void k_attn(
    const __bf16* __restrict__ qac, const __bf16* __restrict__ qbd,
    const __bf16* __restrict__ kh,  const __bf16* __restrict__ kr,
    const float*  __restrict__ amask, float* __restrict__ scores)
{
    int lane = threadIdx.x & 31;
    int w    = threadIdx.x >> 5;
    int bn = blockIdx.z, b = bn / H, n = bn % H;
    int i0 = blockIdx.y * 32 + (w >> 2) * 16;
    int j0 = blockIdx.x * 64 + (w & 3) * 16;
    long actStride = (long)BB * D;
    const __bf16* qa  = qac + (long)b * D + n * DH;
    const __bf16* qb  = qbd + (long)b * D + n * DH;
    const __bf16* kb  = kh  + (long)b * D + n * DH;
    const __bf16* krb = kr  + n * DH;
    int jr0 = j0 - i0 + (Q - 15);   // in [1, 993]
    v8f ac  = {0.f,0.f,0.f,0.f,0.f,0.f,0.f,0.f};
    v8f bd0 = {0.f,0.f,0.f,0.f,0.f,0.f,0.f,0.f};
    v8f bd1 = {0.f,0.f,0.f,0.f,0.f,0.f,0.f,0.f};
#pragma unroll
    for (int k0 = 0; k0 < DH; k0 += 32) {
        v16bf a  = load_fragA(qa, actStride, i0, k0, lane);
        v16bf bk = load_fragB(kb, actStride, j0, k0, lane);
        ac = __builtin_amdgcn_wmma_f32_16x16x32_bf16(false, a, false, bk,
                                                     (short)0, ac, false, false);
        v16bf a2 = load_fragA(qb, actStride, i0, k0, lane);
        v16bf b0 = load_fragB(krb, (long)D, jr0,      k0, lane);
        v16bf b1 = load_fragB(krb, (long)D, jr0 + 16, k0, lane);
        bd0 = __builtin_amdgcn_wmma_f32_16x16x32_bf16(false, a2, false, b0,
                                                      (short)0, bd0, false, false);
        bd1 = __builtin_amdgcn_wmma_f32_16x16x32_bf16(false, a2, false, b1,
                                                      (short)0, bd1, false, false);
    }
    int jl = lane & 15, half = lane >> 4;
    int j  = j0 + jl;
    float amv = amask[b * Q + j];
    bool keymask = (1.f - amv) > 0.f;
#pragma unroll
    for (int r = 0; r < 8; ++r) {
        int i = i0 + half * 8 + r;
        int t = jl - (half * 8 + r) + 15;        // 0..30 : bd column for (i,j)
        int src = (t & 15) | (lane & 16);        // same M-half, column t
        float x0 = bd0[r], x1 = bd1[r];
        float v0 = __shfl(x0, src, 32);
        float v1 = __shfl(x1, src, 32);
        float bdv = (t < 16) ? v0 : v1;
        float sc = (ac[r] + bdv) * SCALE_C;
        if (keymask && (i != j)) sc -= 1e30f;
        scores[((long)bn * Q + i) * Q + j] = sc;
    }
}

// ---------------- softmax over 512 keys, emit bf16 probs ----------------
__global__ __launch_bounds__(256) void k_softmax(const float* __restrict__ scores,
                                                 __bf16* __restrict__ prob) {
    __shared__ float red[256];
    long row = blockIdx.x;
    const float* s = scores + row * Q;
    int tid = threadIdx.x;
    float x0 = s[tid], x1 = s[tid + 256];
    float mx = block_reduce(fmaxf(x0, x1), red, 1);
    float e0 = expf(x0 - mx), e1 = expf(x1 - mx);
    float sum = block_reduce(e0 + e1, red, 0);
    float inv = 1.f / sum;
    prob[row * Q + tid]       = (__bf16)(e0 * inv);
    prob[row * Q + tid + 256] = (__bf16)(e1 * inv);
}

// ---------------- V transpose: vt[bn][d][j] = vh[(j,b)][n*64+d] ----------------
__global__ void k_vt(const float* __restrict__ vh, __bf16* __restrict__ vt) {
    long idx = (long)blockIdx.x * blockDim.x + threadIdx.x;
    if (idx >= (long)BB * H * DH * Q) return;
    int bn = (int)(idx / (DH * Q));
    int rem = (int)(idx % (DH * Q));
    int d = rem / Q, j = rem % Q;
    int b = bn / H, n = bn % H;
    vt[idx] = (__bf16)vh[((long)j * BB + b) * D + n * DH + d];
}

// ---------------- PV: vec[b,n,i,d] = sum_j prob[i,j] * v[j,d] ----------------
__global__ __launch_bounds__(256) void k_pv(const __bf16* __restrict__ prob,
                                            const __bf16* __restrict__ vt,
                                            __bf16* __restrict__ vecbf) {
    int lane = threadIdx.x & 31;
    int w    = threadIdx.x >> 5;
    int bn = blockIdx.z, b = bn / H, n = bn % H;
    int i0 = blockIdx.y * 32 + (w >> 2) * 16;
    int d0 = (w & 3) * 16;
    const __bf16* A  = prob + (long)bn * Q * Q;
    const __bf16* Bt = vt   + (long)bn * DH * Q;
    v8f acc = {0.f,0.f,0.f,0.f,0.f,0.f,0.f,0.f};
    for (int k0 = 0; k0 < Q; k0 += 32) {
        v16bf a  = load_fragA(A,  (long)Q, i0, k0, lane);
        v16bf bf = load_fragB(Bt, (long)Q, d0, k0, lane);
        acc = __builtin_amdgcn_wmma_f32_16x16x32_bf16(false, a, false, bf,
                                                      (short)0, acc, false, false);
    }
    int d = d0 + (lane & 15);
    int rbase = i0 + ((lane >> 4) << 3);
#pragma unroll
    for (int r = 0; r < 8; ++r) {
        int i = rbase + r;
        vecbf[((long)i * BB + b) * D + n * DH + d] = (__bf16)acc[r];
    }
}

// ---------------- residual add + LayerNorm ----------------
__global__ __launch_bounds__(256) void k_addln(const float* __restrict__ x,
                                               float* __restrict__ h, __bf16* __restrict__ hbf,
                                               const float* __restrict__ g,
                                               const float* __restrict__ bv) {
    __shared__ float red[256];
    long t = blockIdx.x;
    int tid = threadIdx.x;
    float v[3]; float s = 0.f;
#pragma unroll
    for (int e = 0; e < 3; ++e) {
        int c = tid + e * 256;
        v[e] = x[t * D + c] + h[t * D + c];
        s += v[e];
    }
    float mean = block_reduce(s, red, 0) / (float)D;
    float s2 = 0.f;
#pragma unroll
    for (int e = 0; e < 3; ++e) { float d = v[e] - mean; s2 += d * d; }
    float var = block_reduce(s2, red, 0) / (float)D;
    float inv = rsqrtf(var + 1e-12f);
#pragma unroll
    for (int e = 0; e < 3; ++e) {
        int c = tid + e * 256;
        float o = (v[e] - mean) * inv * g[c] + bv[c];
        h[t * D + c]   = o;
        hbf[t * D + c] = (__bf16)o;
    }
}

// ---------------- summary head: summ = tanh(last @ sum_w + sum_b) ----------------
__global__ void k_summary(const float* __restrict__ h, const float* __restrict__ sw,
                          const float* __restrict__ sb, float* __restrict__ summ) {
    int idx = blockIdx.x * blockDim.x + threadIdx.x;
    if (idx >= BB * D) return;
    int b = idx / D, jo = idx % D;
    const float* rowp = h + ((long)(Q - 1) * BB + b) * D;
    float s = sb[jo];
    for (int k = 0; k < D; ++k) s += rowp[k] * sw[(long)k * D + jo];
    summ[idx] = tanhf(s);
}

__global__ void k_proj(const float* __restrict__ summ, const float* __restrict__ pw,
                       const float* __restrict__ pb, float* __restrict__ out) {
    int idx = blockIdx.x * blockDim.x + threadIdx.x;
    if (idx >= BB * NLC) return;
    int b = idx / NLC, c = idx % NLC;
    float s = pb[c];
    for (int j = 0; j < D; ++j) s += summ[b * D + j] * pw[(long)j * NLC + c];
    out[idx] = s;
}

// ---------------- host orchestration ----------------
extern "C" void kernel_launch(void* const* d_in, const int* in_sizes, int n_in,
                              void* d_out, int out_size, void* d_ws, size_t ws_size,
                              hipStream_t stream) {
    (void)in_sizes; (void)n_in; (void)out_size; (void)ws_size;
    const int*   tox      = (const int*)  d_in[0];
    const float* amask    = (const float*)d_in[2];
    const float* word_emb = (const float*)d_in[3];
    const float* q_w      = (const float*)d_in[4];
    const float* k_w      = (const float*)d_in[5];
    const float* v_w      = (const float*)d_in[6];
    const float* o_w      = (const float*)d_in[7];
    const float* r_w      = (const float*)d_in[8];
    const float* r_w_bias = (const float*)d_in[9];
    const float* r_r_bias = (const float*)d_in[10];
    const float* ln1_g    = (const float*)d_in[11];
    const float* ln1_b    = (const float*)d_in[12];
    const float* ff_w1    = (const float*)d_in[13];
    const float* ff_b1    = (const float*)d_in[14];
    const float* ff_w2    = (const float*)d_in[15];
    const float* ff_b2    = (const float*)d_in[16];
    const float* ln2_g    = (const float*)d_in[17];
    const float* ln2_b    = (const float*)d_in[18];
    const float* sum_w    = (const float*)d_in[19];
    const float* sum_b    = (const float*)d_in[20];
    const float* proj_w   = (const float*)d_in[21];
    const float* proj_b   = (const float*)d_in[22];
    float* out = (float*)d_out;

    char* wsp = (char*)d_ws;
    size_t off = 0;
    auto alloc = [&](size_t bytes) -> void* {
        void* p = wsp + off;
        off += (bytes + 255) & ~(size_t)255;
        return p;
    };
    const size_t WTE = (size_t)LNUM * D * D;     // per-weight-set elems
    const size_t WFE = (size_t)LNUM * D * FF;
    __bf16* wqT  = (__bf16*)alloc(WTE * 2);
    __bf16* wkT  = (__bf16*)alloc(WTE * 2);
    __bf16* wvT  = (__bf16*)alloc(WTE * 2);
    __bf16* wrT  = (__bf16*)alloc(WTE * 2);
    __bf16* woB  = (__bf16*)alloc(WTE * 2);      // o_w is already B^T layout
    __bf16* w1T  = (__bf16*)alloc(WFE * 2);      // [FF][D]
    __bf16* w2T  = (__bf16*)alloc(WFE * 2);      // [D][FF]
    __bf16* rbf  = (__bf16*)alloc((size_t)1024 * D * 2);
    __bf16* krbf = (__bf16*)alloc((size_t)KRROWS * D * 2);
    float*  hF   = (float*) alloc((size_t)NT * D * 4);
    __bf16* hbf  = (__bf16*)alloc((size_t)NT * D * 2);
    float*  qh   = (float*) alloc((size_t)NT * D * 4);
    __bf16* qac  = (__bf16*)alloc((size_t)NT * D * 2);
    __bf16* qbd  = (__bf16*)alloc((size_t)NT * D * 2);
    __bf16* khbf = (__bf16*)alloc((size_t)NT * D * 2);
    float*  vhF  = (float*) alloc((size_t)NT * D * 4);
    __bf16* vt   = (__bf16*)alloc((size_t)BB * H * DH * Q * 2);
    float*  scor = (float*) alloc((size_t)BB * H * Q * Q * 4);
    __bf16* prob = (__bf16*)alloc((size_t)BB * H * Q * Q * 2);
    __bf16* vec  = (__bf16*)alloc((size_t)NT * D * 2);
    float*  attn = (float*) alloc((size_t)NT * D * 4);
    __bf16* ffa  = (__bf16*)alloc((size_t)NT * FF * 2);
    float*  ffo  = (float*) alloc((size_t)NT * D * 4);
    float*  summ = (float*) alloc((size_t)BB * D * 4);

    dim3 tb(32, 8);
    // weight conversions (bf16, transposed where B^T needed)
    k_convT<<<dim3(D / 32, D / 32, LNUM), tb, 0, stream>>>(q_w, wqT, D, D, (long)D * D, (long)D * D);
    k_convT<<<dim3(D / 32, D / 32, LNUM), tb, 0, stream>>>(k_w, wkT, D, D, (long)D * D, (long)D * D);
    k_convT<<<dim3(D / 32, D / 32, LNUM), tb, 0, stream>>>(v_w, wvT, D, D, (long)D * D, (long)D * D);
    k_convT<<<dim3(D / 32, D / 32, LNUM), tb, 0, stream>>>(r_w, wrT, D, D, (long)D * D, (long)D * D);
    k_conv<<<4096, 256, 0, stream>>>(o_w, woB, (long)WTE);
    k_convT<<<dim3(FF / 32, D / 32, LNUM), tb, 0, stream>>>(ff_w1, w1T, D, FF, (long)D * FF, (long)D * FF);
    k_convT<<<dim3(D / 32, FF / 32, LNUM), tb, 0, stream>>>(ff_w2, w2T, FF, D, (long)D * FF, (long)D * FF);

    (void)hipMemsetAsync(krbf, 0, (size_t)KRROWS * D * 2, stream);
    k_embed<<<NT, 256, 0, stream>>>(tox, word_emb, hF, hbf);
    k_posenc<<<1024, 256, 0, stream>>>(rbf);

    for (int l = 0; l < LNUM; ++l) {
        const __bf16* WQ = wqT + (size_t)l * D * D;
        const __bf16* WK = wkT + (size_t)l * D * D;
        const __bf16* WV = wvT + (size_t)l * D * D;
        const __bf16* WR = wrT + (size_t)l * D * D;
        const __bf16* WO = woB + (size_t)l * D * D;
        const __bf16* W1 = w1T + (size_t)l * D * FF;
        const __bf16* W2 = w2T + (size_t)l * D * FF;

        k_gemm<<<dim3(D / TN, NT / TM), 256, 0, stream>>>(hbf, D, WQ, D, qh, nullptr, D,
                                                          NT, D, D, nullptr, GF_F32);
        k_qbias<<<(NT * D + 255) / 256, 256, 0, stream>>>(qh, r_w_bias + l * D,
                                                          r_r_bias + l * D, qac, qbd);
        k_gemm<<<dim3(D / TN, NT / TM), 256, 0, stream>>>(hbf, D, WK, D, nullptr, khbf, D,
                                                          NT, D, D, nullptr, GF_BF16);
        k_gemm<<<dim3(D / TN, NT / TM), 256, 0, stream>>>(hbf, D, WV, D, vhF, nullptr, D,
                                                          NT, D, D, nullptr, GF_F32);
        k_gemm<<<dim3(D / TN, 1024 / TM), 256, 0, stream>>>(rbf, D, WR, D, nullptr, krbf, D,
                                                            1024, D, D, nullptr, GF_BF16);
        k_vt<<<((long)BB * H * DH * Q + 255) / 256, 256, 0, stream>>>(vhF, vt);
        k_attn<<<dim3(Q / 64, Q / 32, BB * H), 256, 0, stream>>>(qac, qbd, khbf, krbf,
                                                                 amask, scor);
        k_softmax<<<BB * H * Q, 256, 0, stream>>>(scor, prob);
        k_pv<<<dim3(1, Q / 32, BB * H), 256, 0, stream>>>(prob, vt, vec);
        k_gemm<<<dim3(D / TN, NT / TM), 256, 0, stream>>>(vec, D, WO, D, attn, nullptr, D,
                                                          NT, D, D, nullptr, GF_F32);
        k_addln<<<NT, 256, 0, stream>>>(attn, hF, hbf, ln1_g + l * D, ln1_b + l * D);
        k_gemm<<<dim3(FF / TN, NT / TM), 256, 0, stream>>>(hbf, D, W1, D, nullptr, ffa, FF,
                                                           NT, FF, D, ff_b1 + l * FF,
                                                           GF_BF16 | GF_BIAS | GF_GELU);
        k_gemm<<<dim3(D / TN, NT / TM), 256, 0, stream>>>(ffa, FF, W2, FF, ffo, nullptr, D,
                                                          NT, D, FF, ff_b2 + l * D,
                                                          GF_F32 | GF_BIAS);
        k_addln<<<NT, 256, 0, stream>>>(ffo, hF, hbf, ln2_g + l * D, ln2_b + l * D);
    }

    k_summary<<<(BB * D + 255) / 256, 256, 0, stream>>>(hF, sum_w, sum_b, summ);
    k_proj<<<1, 64, 0, stream>>>(summ, proj_w, proj_b, out);
}
